// TopologicalContrastiveLoss_81003083202859
// MI455X (gfx1250) — compile-verified, matrix operand order checked by reading, not run on hardware
//
#include <hip/hip_runtime.h>
#include <hip/hip_bf16.h>
#include <math.h>

// TopologicalContrastiveLoss on MI455X (gfx1250, wave32)
// Pipeline:
//   1) split_kernel : fp32 -> bf16 hi/lo decomposition (bf16x3 scheme) + row sqnorms
//   2) dist_gemm    : WMMA bf16 GEMM hot loop (pure loads + v_wmma, no conversions)
//   3) prim_kernel  : MST deaths (1 workgroup / matrix, register-resident frontier)
//   4) sort_kernel  : bitonic sort of deaths
//   5) loss_kernel  : W1 distances + triplet margin
// Workspace requirement: ~243 MB.

#define N_PTS 4096
#define DIMS  1024

typedef __attribute__((ext_vector_type(16))) __bf16 v16bf;
typedef __attribute__((ext_vector_type(8)))  __bf16 v8bf;
typedef __attribute__((ext_vector_type(8)))  float  v8f;

__device__ __forceinline__ v8f wmma_bf16(v16bf a, v16bf b, v8f c) {
  return __builtin_amdgcn_wmma_f32_16x16x32_bf16(
      /*neg_a=*/false, a, /*neg_b=*/false, b,
      /*c_mod=*/(short)0, c, /*reuse_a=*/false, /*reuse_b=*/false);
}

// Load a 16-element bf16 fragment: per-lane k-slices [kb,kb+8) and [kb+16,kb+24)
// are contiguous in the row-major bf16 array -> two aligned global_load_b128.
__device__ __forceinline__ v16bf load_frag16(const __bf16* __restrict__ row, int kb) {
  union { v16bf v; struct { v8bf a, b; } s; } u;
  u.s.a = *(const v8bf*)(row + kb);
  u.s.b = *(const v8bf*)(row + kb + 16);
  return u.v;
}

// ---------------------------------------------------------------------------
// Kernel 1: bf16 hi/lo split + per-row squared norms (one wave per row).
// Each lane owns 32 contiguous elements of its row.
// ---------------------------------------------------------------------------
__global__ __launch_bounds__(256) void split_kernel(
    const float* __restrict__ X0, const float* __restrict__ X1,
    const float* __restrict__ X2, __bf16* __restrict__ Xhi,
    __bf16* __restrict__ Xlo, float* __restrict__ sq) {
  int gw = (int)((blockIdx.x * blockDim.x + threadIdx.x) >> 5);
  int lane = threadIdx.x & 31;
  int z = gw / N_PTS;
  int row = gw % N_PTS;
  if (z >= 3) return;
  const float* X = (z == 0) ? X0 : (z == 1) ? X1 : X2;
  const float* r = X + (size_t)row * DIMS;
  __bf16* oh = Xhi + ((size_t)z * N_PTS + row) * DIMS;
  __bf16* ol = Xlo + ((size_t)z * N_PTS + row) * DIMS;

  const int base = lane * 32;
  float s = 0.f;
#pragma unroll
  for (int c = 0; c < 32; c += 8) {
    float4 f0 = *(const float4*)(r + base + c);
    float4 f1 = *(const float4*)(r + base + c + 4);
    float v[8] = {f0.x, f0.y, f0.z, f0.w, f1.x, f1.y, f1.z, f1.w};
    v8bf h, l;
#pragma unroll
    for (int e = 0; e < 8; ++e) {
      s += v[e] * v[e];
      __bf16 hh = (__bf16)v[e];
      h[e] = hh;
      l[e] = (__bf16)(v[e] - (float)hh);
    }
    *(v8bf*)(oh + base + c) = h;
    *(v8bf*)(ol + base + c) = l;
  }
#pragma unroll
  for (int off = 16; off; off >>= 1) s += __shfl_xor(s, off, 32);
  if (lane == 0) sq[z * N_PTS + row] = s;
}

// ---------------------------------------------------------------------------
// Kernel 2: distance GEMM via WMMA (bf16x3: hi*hi + hi*lo + lo*hi).
// Block = 256 threads = 8 waves; each wave computes a 16x64 strip (4 tiles),
// block covers 128x64. Hot loop: 20 b128 loads + 12 v_wmma per k-tile.
// D[i][j] = sqrt(max(sq_i + sq_j - 2*dot, 1e-12)).
// ---------------------------------------------------------------------------
__global__ __launch_bounds__(256) void dist_gemm_kernel(
    const __bf16* __restrict__ Xhi, const __bf16* __restrict__ Xlo,
    const float* __restrict__ sq, float* __restrict__ dist) {
  const int z = blockIdx.z;
  const __bf16* Hz = Xhi + (size_t)z * N_PTS * DIMS;
  const __bf16* Lz = Xlo + (size_t)z * N_PTS * DIMS;
  const float* sqz = sq + z * N_PTS;
  float* Dz = dist + (size_t)z * N_PTS * N_PTS;

  const int wave = threadIdx.x >> 5;
  const int lane = threadIdx.x & 31;
  const int lm = lane & 15;
  const int base0 = (lane & 16) ? 8 : 0;   // ISA 7.12.2 half-wave K offset

  const int i0 = blockIdx.x * 128 + wave * 16;
  const int j0 = blockIdx.y * 64;

  const size_t arow = (size_t)(i0 + lm) * DIMS;
  const size_t brow0 = (size_t)(j0 +  0 + lm) * DIMS;
  const size_t brow1 = (size_t)(j0 + 16 + lm) * DIMS;
  const size_t brow2 = (size_t)(j0 + 32 + lm) * DIMS;
  const size_t brow3 = (size_t)(j0 + 48 + lm) * DIMS;

  v8f c0 = {}, c1 = {}, c2 = {}, c3 = {};

  for (int kt = 0; kt < DIMS / 32; ++kt) {
    const int kb = kt * 32 + base0;
    v16bf ah = load_frag16(Hz + arow, kb);
    v16bf al = load_frag16(Lz + arow, kb);

    v16bf bh = load_frag16(Hz + brow0, kb);
    v16bf bl = load_frag16(Lz + brow0, kb);
    c0 = wmma_bf16(ah, bh, c0);
    c0 = wmma_bf16(ah, bl, c0);
    c0 = wmma_bf16(al, bh, c0);

    bh = load_frag16(Hz + brow1, kb);
    bl = load_frag16(Lz + brow1, kb);
    c1 = wmma_bf16(ah, bh, c1);
    c1 = wmma_bf16(ah, bl, c1);
    c1 = wmma_bf16(al, bh, c1);

    bh = load_frag16(Hz + brow2, kb);
    bl = load_frag16(Lz + brow2, kb);
    c2 = wmma_bf16(ah, bh, c2);
    c2 = wmma_bf16(ah, bl, c2);
    c2 = wmma_bf16(al, bh, c2);

    bh = load_frag16(Hz + brow3, kb);
    bl = load_frag16(Lz + brow3, kb);
    c3 = wmma_bf16(ah, bh, c3);
    c3 = wmma_bf16(ah, bl, c3);
    c3 = wmma_bf16(al, bh, c3);
  }

  // Epilogue: C/D layout — VGPR r, lanes 0-15: M=r, N=lane; lanes 16-31: M=8+r.
  const int mbase = (lane & 16) ? 8 : 0;
#pragma unroll
  for (int r = 0; r < 8; ++r) {
    const int row = i0 + mbase + r;
    const float sqi = sqz[row];
    float* outrow = Dz + (size_t)row * N_PTS;
    {
      int col = j0 + lm;
      float d2 = sqi + sqz[col] - 2.0f * c0[r];
      outrow[col] = sqrtf(fmaxf(d2, 1e-12f));
    }
    {
      int col = j0 + 16 + lm;
      float d2 = sqi + sqz[col] - 2.0f * c1[r];
      outrow[col] = sqrtf(fmaxf(d2, 1e-12f));
    }
    {
      int col = j0 + 32 + lm;
      float d2 = sqi + sqz[col] - 2.0f * c2[r];
      outrow[col] = sqrtf(fmaxf(d2, 1e-12f));
    }
    {
      int col = j0 + 48 + lm;
      float d2 = sqi + sqz[col] - 2.0f * c3[r];
      outrow[col] = sqrtf(fmaxf(d2, 1e-12f));
    }
  }
}

// ---------------------------------------------------------------------------
// Kernel 3: Prim MST (one 1024-thread workgroup per distance matrix).
// Each thread owns 4 columns in registers; hierarchical argmin:
// wave shuffle reduce -> LDS 32 -> wave0 reduce. 2 barriers / iteration.
// Key = (float_bits << 32) | col  (ties -> lowest index, matches argmin).
// ---------------------------------------------------------------------------
__global__ __launch_bounds__(1024) void prim_kernel(
    const float* __restrict__ dist, float* __restrict__ deaths) {
  const int z = blockIdx.x;
  const float* D = dist + (size_t)z * N_PTS * N_PTS;
  float* out = deaths + z * N_PTS;

  __shared__ unsigned long long warpmin[33];

  const int tid = threadIdx.x;
  const int lane = tid & 31;
  const int wid = tid >> 5;

  float mind[4];
  bool vis[4];
#pragma unroll
  for (int t = 0; t < 4; ++t) {
    int c = tid + t * 1024;
    mind[t] = D[c];          // distances from vertex 0
    vis[t] = (c == 0);       // vertex 0 starts visited
  }

  for (int it = 0; it < N_PTS - 1; ++it) {
    unsigned long long best = 0xFFFFFFFFFFFFFFFFull;
#pragma unroll
    for (int t = 0; t < 4; ++t) {
      if (!vis[t]) {
        unsigned int fb = __float_as_uint(mind[t]);  // positive -> monotonic
        unsigned long long key =
            ((unsigned long long)fb << 32) | (unsigned)(tid + t * 1024);
        best = (key < best) ? key : best;
      }
    }
#pragma unroll
    for (int off = 16; off; off >>= 1) {
      unsigned long long o = __shfl_xor(best, off, 32);
      best = (o < best) ? o : best;
    }
    if (lane == 0) warpmin[wid] = best;
    __syncthreads();
    if (wid == 0) {
      unsigned long long b = warpmin[lane];
#pragma unroll
      for (int off = 16; off; off >>= 1) {
        unsigned long long o = __shfl_xor(b, off, 32);
        b = (o < b) ? o : b;
      }
      if (lane == 0) warpmin[32] = b;
    }
    __syncthreads();
    const unsigned long long win = warpmin[32];
    const unsigned int j = (unsigned)(win & 0xFFFFFFFFu);
    if (tid == 0) out[it] = __uint_as_float((unsigned)(win >> 32));
    if ((int)(j & 1023) == tid) vis[j >> 10] = true;  // owner marks visited
    const float* rowj = D + (size_t)j * N_PTS;
#pragma unroll
    for (int t = 0; t < 4; ++t) {
      float d = rowj[tid + t * 1024];   // coalesced, L2-resident
      mind[t] = fminf(mind[t], d);
    }
  }
}

// ---------------------------------------------------------------------------
// Kernel 4: bitonic sort of 4095 deaths (pad slot -> +inf), one WG per array.
// ---------------------------------------------------------------------------
__global__ __launch_bounds__(1024) void sort_kernel(
    const float* __restrict__ deaths, float* __restrict__ sorted) {
  const int z = blockIdx.x;
  __shared__ float s[4096];
  const int tid = threadIdx.x;
#pragma unroll
  for (int t = 0; t < 4; ++t) {
    int i = tid + t * 1024;
    s[i] = (i < N_PTS - 1) ? deaths[z * N_PTS + i] : 3.0e38f;
  }
  __syncthreads();
  for (int k = 2; k <= 4096; k <<= 1) {
    for (int j = k >> 1; j > 0; j >>= 1) {
#pragma unroll
      for (int t = 0; t < 4; ++t) {
        int p = tid + t * 1024;
        int q = p ^ j;
        if (q > p) {
          float a = s[p], b = s[q];
          bool up = ((p & k) == 0);
          if ((a > b) == up) { s[p] = b; s[q] = a; }
        }
      }
      __syncthreads();
    }
  }
#pragma unroll
  for (int t = 0; t < 4; ++t) {
    int i = tid + t * 1024;
    if (i < N_PTS - 1) sorted[z * N_PTS + i] = s[i];
  }
}

// ---------------------------------------------------------------------------
// Kernel 5: W1 distances + triplet margin.
// ---------------------------------------------------------------------------
__global__ __launch_bounds__(1024) void loss_kernel(
    const float* __restrict__ sorted, float* __restrict__ out) {
  __shared__ float redp[32];
  __shared__ float redn[32];
  const int tid = threadIdx.x;
  const int lane = tid & 31;
  const int wid = tid >> 5;
  const float* sa = sorted;
  const float* sp = sorted + N_PTS;
  const float* sn = sorted + 2 * N_PTS;
  float pos = 0.f, neg = 0.f;
  for (int i = tid; i < N_PTS - 1; i += 1024) {
    float a = sa[i];
    pos += fabsf(a - sp[i]);
    neg += fabsf(a - sn[i]);
  }
#pragma unroll
  for (int off = 16; off; off >>= 1) {
    pos += __shfl_xor(pos, off, 32);
    neg += __shfl_xor(neg, off, 32);
  }
  if (lane == 0) { redp[wid] = pos; redn[wid] = neg; }
  __syncthreads();
  if (tid == 0) {
    float P = 0.f, Ng = 0.f;
    for (int w = 0; w < 32; ++w) { P += redp[w]; Ng += redn[w]; }
    out[0] = fmaxf(P - Ng + 1.0f, 0.0f);
  }
}

// ---------------------------------------------------------------------------
extern "C" void kernel_launch(void* const* d_in, const int* in_sizes, int n_in,
                              void* d_out, int out_size, void* d_ws, size_t ws_size,
                              hipStream_t stream) {
  (void)in_sizes; (void)n_in; (void)out_size; (void)ws_size;
  const float* A  = (const float*)d_in[0];
  const float* P  = (const float*)d_in[1];
  const float* Ng = (const float*)d_in[2];

  char* wsb = (char*)d_ws;
  float* sq     = (float*)wsb;  wsb += (size_t)3 * N_PTS * sizeof(float);
  float* deaths = (float*)wsb;  wsb += (size_t)3 * N_PTS * sizeof(float);
  float* sorted = (float*)wsb;  wsb += (size_t)3 * N_PTS * sizeof(float);
  __bf16* Xhi   = (__bf16*)wsb; wsb += (size_t)3 * N_PTS * DIMS * sizeof(__bf16);
  __bf16* Xlo   = (__bf16*)wsb; wsb += (size_t)3 * N_PTS * DIMS * sizeof(__bf16);
  float* dist   = (float*)wsb;  // 3 * N_PTS * N_PTS floats (~192 MB)

  split_kernel<<<dim3(3 * N_PTS / 8), 256, 0, stream>>>(A, P, Ng, Xhi, Xlo, sq);
  dist_gemm_kernel<<<dim3(N_PTS / 128, N_PTS / 64, 3), 256, 0, stream>>>(
      Xhi, Xlo, sq, dist);
  prim_kernel<<<dim3(3), 1024, 0, stream>>>(dist, deaths);
  sort_kernel<<<dim3(3), 1024, 0, stream>>>(deaths, sorted);
  loss_kernel<<<dim3(1), 1024, 0, stream>>>(sorted, (float*)d_out);
}
